// GPDT_V2_28192165331061
// MI455X (gfx1250) — matrix operand order repaired and checked by reading, not compiled, not run on hardware
//
#include <hip/hip_runtime.h>
#include <hip/hip_bf16.h>

#define B_   4
#define L_   128
#define NN   9
#define NPB  (L_ * NN)      // 1152 nodes per batch
#define HH   128
#define MTOT (B_ * NPB)     // 4608 rows for all batched GEMMs

typedef __attribute__((ext_vector_type(16))) _Float16 v16h;
typedef __attribute__((ext_vector_type(8)))  _Float16 v8h;
typedef __attribute__((ext_vector_type(8)))  float    v8f;

// incoming structural neighbors (bidirectional tree edges), analytic from reference builder
__constant__ int c_deg[NN]     = {3,2,2,1,2,2,1,2,1};
__constant__ int c_nbr[NN][3]  = {{1,4,7},{0,2},{1,3},{2,0,0},{0,5},{4,6},{5,0,0},{0,8},{7,0,0}};

// ---------------------------------------------------------------------------
// Weight pack: W (K x P, f32 row-major) -> f16 WMMA B-fragments.
// Fragment (nt, kt): 32 lanes x 16 contiguous f16.
//   lane: ncol = lane&15, khi = lane>>4 ; element j -> k = kt*32 + (j>>3)*16 + khi*8 + (j&7)
// Wp[((nt*(K/32)+kt)*32 + lane)*16 + j] = W[k*P + nt*16 + ncol]
// ---------------------------------------------------------------------------
__global__ __launch_bounds__(256)
void pack_w_kernel(const float* __restrict__ W, _Float16* __restrict__ Wp, int K, int P)
{
    const int idx  = blockIdx.x * 256 + threadIdx.x;   // one per element, total K*P
    const int j    = idx & 15;
    const int lane = (idx >> 4) & 31;
    const int tile = idx >> 9;
    const int ktiles = K >> 5;
    const int kt = tile % ktiles;
    const int nt = tile / ktiles;
    const int khi = lane >> 4;
    const int col = nt * 16 + (lane & 15);
    const int k   = kt * 32 + (j >> 3) * 16 + khi * 8 + (j & 7);
    Wp[idx] = (_Float16)W[(size_t)k * P + col];
}

// ---------------------------------------------------------------------------
// WMMA GEMM: C = op(A16[M x K] @ W[K x P] + bias)
//   A16: f16 row-major activations; Wp: pre-packed f16 fragments.
//   OP: 0 = store f32 ; 1 = relu, store f16 only (C16) ; 2 = +residual, store f32
// block = 256 thr (8 waves); wave owns one 16x16 tile; block = 16(M) x 128(P).
// grid = (M/16, P/128).
// ---------------------------------------------------------------------------
template<int OP, int K, int P>
__global__ __launch_bounds__(256)
void gemm_wmma(const _Float16* __restrict__ A, const _Float16* __restrict__ Wp,
               const float* __restrict__ bias, const float* __restrict__ R,
               float* __restrict__ C, _Float16* __restrict__ C16)
{
    constexpr int KT = K / 32;
    const int wave = threadIdx.x >> 5;
    const int lane = threadIdx.x & 31;
    const int mo   = blockIdx.x * 16;
    const int nt   = blockIdx.y * 8 + wave;
    const int no   = nt * 16;
    const int mrow = lane & 15;
    const int khi  = lane >> 4;
    const int ncol = lane & 15;

    const float bv = bias[no + ncol];
    v8f acc;
#pragma unroll
    for (int r = 0; r < 8; ++r) acc[r] = bv;

    const _Float16* arow = A + (size_t)(mo + mrow) * K + khi * 8;
    const _Float16* bptr = Wp + (size_t)nt * KT * 512 + (size_t)lane * 16;

#pragma unroll
    for (int kt = 0; kt < KT; ++kt) {
        // A fragment: two aligned 16B chunks (K j<8 and j>=8 halves)
        v8h alo = *(const v8h*)(arow + kt * 32);
        v8h ahi = *(const v8h*)(arow + kt * 32 + 16);
        v16h a = __builtin_shufflevector(alo, ahi,
                    0,1,2,3,4,5,6,7,8,9,10,11,12,13,14,15);
        // B fragment: 32 contiguous bytes, pre-packed
        v16h b = *(const v16h*)(bptr + (size_t)kt * 512);

        acc = __builtin_amdgcn_wmma_f32_16x16x32_f16(
            false, a, false, b, (short)0, acc, false, false);
    }

#pragma unroll
    for (int r = 0; r < 8; ++r) {
        const int row = mo + khi * 8 + r;
        const int col = no + ncol;
        float v = acc[r];
        if (OP == 1) {
            v = v > 0.f ? v : 0.f;
            C16[(size_t)row * P + col] = (_Float16)v;
        } else {
            if (OP == 2) v += R[(size_t)row * P + col];
            C[(size_t)row * P + col] = v;
        }
    }
}

// ---------------------------------------------------------------------------
// Graph attention, analytic edges. One block (128 thr, 4 waves) per dst node.
// Sources: (t,i) for t<=l  plus  (l,j) for j in nbr(i).  nsrc <= 131.
// Output (f16): attn = softmax-agg(V) + skip   (skip = SKP f32, bias included)
// ---------------------------------------------------------------------------
__global__ __launch_bounds__(128)
void attn_kernel(const float* __restrict__ Q, const float* __restrict__ Km,
                 const float* __restrict__ V, const float* __restrict__ SKP,
                 _Float16* __restrict__ OUT16)
{
    const int node = blockIdx.x;                 // b*NPB + l*9 + i
    const int b = node / NPB;
    const int ln = node % NPB;
    const int l = ln / NN;
    const int i = ln % NN;
    const int tid = threadIdx.x;
    const int wave = tid >> 5, lane = tid & 31;

    __shared__ float s_q[HH];
    __shared__ float s_sc[136];
    __shared__ float s_m[4], s_s[4];

    s_q[tid] = Q[(size_t)node * HH + tid];
    __syncthreads();

    const int nsrc = (l + 1) + c_deg[i];

    // scores: one wave per source (strided)
    for (int s = wave; s < nsrc; s += 4) {
        int srow;
        if (s <= l) srow = b * NPB + s * NN + i;
        else        srow = b * NPB + l * NN + c_nbr[i][s - l - 1];
        const float* kr = Km + (size_t)srow * HH;
        float p = 0.f;
        for (int j = lane; j < HH; j += 32) p += s_q[j] * kr[j];
        for (int off = 16; off > 0; off >>= 1) p += __shfl_xor(p, off, 32);
        if (lane == 0) s_sc[s] = p * 0.08838834764831845f;   // 1/sqrt(128)
    }
    __syncthreads();

    // softmax max
    float mx = -1e30f;
    for (int s = tid; s < nsrc; s += 128) mx = fmaxf(mx, s_sc[s]);
    for (int off = 16; off > 0; off >>= 1) mx = fmaxf(mx, __shfl_xor(mx, off, 32));
    if (lane == 0) s_m[wave] = mx;
    __syncthreads();
    mx = fmaxf(fmaxf(s_m[0], s_m[1]), fmaxf(s_m[2], s_m[3]));

    // exp + sum
    float sm = 0.f;
    for (int s = tid; s < nsrc; s += 128) {
        float e = __expf(s_sc[s] - mx);
        s_sc[s] = e;
        sm += e;
    }
    for (int off = 16; off > 0; off >>= 1) sm += __shfl_xor(sm, off, 32);
    if (lane == 0) s_s[wave] = sm;
    __syncthreads();
    const float inv = 1.f / (s_s[0] + s_s[1] + s_s[2] + s_s[3]);

    // weighted aggregation of V; thread = feature channel
    float acc = 0.f;
    for (int s = 0; s < nsrc; ++s) {
        int srow;
        if (s <= l) srow = b * NPB + s * NN + i;
        else        srow = b * NPB + l * NN + c_nbr[i][s - l - 1];
        acc += s_sc[s] * V[(size_t)srow * HH + tid];
    }
    const size_t o = (size_t)node * HH + tid;
    OUT16[o] = (_Float16)(acc * inv + SKP[o]);
}

// ---------------------------------------------------------------------------
// Row LayerNorm (128 features). One block per row; in-place safe.
// Writes f32 (residual/consumer path) + f16 copy (GEMM A-operand path).
// ---------------------------------------------------------------------------
__global__ __launch_bounds__(128)
void ln_kernel(const float* __restrict__ X, const float* __restrict__ g,
               const float* __restrict__ bt, float* __restrict__ O,
               _Float16* __restrict__ O16)
{
    const int row = blockIdx.x, tid = threadIdx.x;
    const int wave = tid >> 5, lane = tid & 31;
    __shared__ float s1[4], s2[4];

    float v = X[(size_t)row * HH + tid];
    float sm = v;
    for (int off = 16; off > 0; off >>= 1) sm += __shfl_xor(sm, off, 32);
    if (lane == 0) s1[wave] = sm;
    __syncthreads();
    const float mean = (s1[0] + s1[1] + s1[2] + s1[3]) * (1.f / HH);
    const float d = v - mean;
    float vs = d * d;
    for (int off = 16; off > 0; off >>= 1) vs += __shfl_xor(vs, off, 32);
    if (lane == 0) s2[wave] = vs;
    __syncthreads();
    const float var = (s2[0] + s2[1] + s2[2] + s2[3]) * (1.f / HH);
    const float r = d * rsqrtf(var + 1e-5f) * g[tid] + bt[tid];
    O[(size_t)row * HH + tid]   = r;
    O16[(size_t)row * HH + tid] = (_Float16)r;
}

// ---------------------------------------------------------------------------
// Embedding (root / nonroot / return / time) + input LayerNorm -> X, X16
// ---------------------------------------------------------------------------
__global__ __launch_bounds__(128)
void embed_ln_kernel(const float* __restrict__ states, const float* __restrict__ actions,
                     const float* __restrict__ rtg, const int* __restrict__ tsteps,
                     const float* __restrict__ rw, const float* __restrict__ rb,
                     const float* __restrict__ nw, const float* __restrict__ nb,
                     const float* __restrict__ retw, const float* __restrict__ retb,
                     const float* __restrict__ tw,
                     const float* __restrict__ lg, const float* __restrict__ lb,
                     float* __restrict__ X, _Float16* __restrict__ X16)
{
    const int node = blockIdx.x;
    const int b = node / NPB;
    const int ln = node % NPB;
    const int l = ln / NN;
    const int i = ln % NN;
    const int h = threadIdx.x;
    const int wave = h >> 5, lane = h & 31;
    __shared__ float s1[4], s2[4];

    const float* st = states + ((size_t)b * L_ + l) * 27;
    float e;
    if (i == 0) {
        e = rb[h];
#pragma unroll
        for (int s = 0; s < 11; ++s) e += st[s] * rw[s * HH + h];
    } else {
        const int j = i - 1;
        e = nb[h]
          + st[11 + 2 * j] * nw[h]
          + st[12 + 2 * j] * nw[HH + h]
          + actions[((size_t)b * (L_ + 1) + l) * 8 + j] * nw[2 * HH + h];
    }
    e += rtg[(size_t)b * L_ + l] * retw[h] + retb[h];
    e += tw[(size_t)tsteps[(size_t)b * L_ + l] * HH + h];

    float sm = e;
    for (int off = 16; off > 0; off >>= 1) sm += __shfl_xor(sm, off, 32);
    if (lane == 0) s1[wave] = sm;
    __syncthreads();
    const float mean = (s1[0] + s1[1] + s1[2] + s1[3]) * (1.f / HH);
    const float d = e - mean;
    float vs = d * d;
    for (int off = 16; off > 0; off >>= 1) vs += __shfl_xor(vs, off, 32);
    if (lane == 0) s2[wave] = vs;
    __syncthreads();
    const float var = (s2[0] + s2[1] + s2[2] + s2[3]) * (1.f / HH);
    const float r = d * rsqrtf(var + 1e-5f) * lg[h] + lb[h];
    X[(size_t)node * HH + h]   = r;
    X16[(size_t)node * HH + h] = (_Float16)r;
}

// ---------------------------------------------------------------------------
// preds = tanh(x[:, :, 1:, :] @ predict_w + predict_b)  -> (B, L, 8, 1)
// ---------------------------------------------------------------------------
__global__ __launch_bounds__(128)
void predict_kernel(const float* __restrict__ X, const float* __restrict__ pw,
                    const float* __restrict__ pb, float* __restrict__ out)
{
    const int idx = blockIdx.x;            // b*L*8 + l*8 + i
    const int i  = idx % 8;
    const int bl = idx / 8;
    const int l  = bl % L_;
    const int b  = bl / L_;
    const int row = b * NPB + l * NN + (i + 1);
    const int tid = threadIdx.x;
    __shared__ float s[4];

    float p = X[(size_t)row * HH + tid] * pw[tid];
    for (int off = 16; off > 0; off >>= 1) p += __shfl_xor(p, off, 32);
    if ((tid & 31) == 0) s[tid >> 5] = p;
    __syncthreads();
    if (tid == 0) out[idx] = tanhf(s[0] + s[1] + s[2] + s[3] + pb[0]);
}

// ---------------------------------------------------------------------------
// Host launch
// ---------------------------------------------------------------------------
struct ParamReader {
    void* const* din; int idx; const float* base; size_t off; bool concat;
    const float* next(size_t count) {
        if (concat) { const float* p = base + off; off += count; return p; }
        return (const float*)din[idx++];
    }
};

extern "C" void kernel_launch(void* const* d_in, const int* in_sizes, int n_in,
                              void* d_out, int out_size, void* d_ws, size_t ws_size,
                              hipStream_t stream)
{
    const float* states  = (const float*)d_in[0];
    const float* actions = (const float*)d_in[1];
    /* d_in[2] rewards: unused by reference */
    const float* rtg     = (const float*)d_in[3];
    const int*   tsteps  = (const int*)d_in[4];
    /* d_in[5]/d_in[6] edge arrays: adjacency reproduced analytically */

    ParamReader pr;
    pr.din = d_in; pr.idx = 7; pr.off = 0;
    pr.concat = (n_in == 8);
    pr.base = pr.concat ? (const float*)d_in[7] : nullptr;

    const float* e_root_w = pr.next(11 * HH);
    const float* e_root_b = pr.next(HH);
    const float* e_nr_w   = pr.next(3 * HH);
    const float* e_nr_b   = pr.next(HH);
    const float* e_ret_w  = pr.next(HH);
    const float* e_ret_b  = pr.next(HH);
    const float* e_time   = pr.next(1000 * HH);
    const float* ln_g     = pr.next(HH);
    const float* ln_b     = pr.next(HH);
    const float* pred_w   = pr.next(HH);
    const float* pred_b   = pr.next(1);

    struct LayerP {
        const float *q_w,*q_b,*k_w,*k_b,*v_w,*v_b,*skip_w,*skip_b,*O_w,*O_b,
                    *ln1_g,*ln1_b,*ffn1_w,*ffn1_b,*ffn2_w,*ffn2_b,*ln2_g,*ln2_b;
    } lp[3];
    for (int li = 0; li < 3; ++li) {
        lp[li].q_w    = pr.next(HH*HH);  lp[li].q_b    = pr.next(HH);
        lp[li].k_w    = pr.next(HH*HH);  lp[li].k_b    = pr.next(HH);
        lp[li].v_w    = pr.next(HH*HH);  lp[li].v_b    = pr.next(HH);
        lp[li].skip_w = pr.next(HH*HH);  lp[li].skip_b = pr.next(HH);
        lp[li].O_w    = pr.next(HH*HH);  lp[li].O_b    = pr.next(HH);
        lp[li].ln1_g  = pr.next(HH);     lp[li].ln1_b  = pr.next(HH);
        lp[li].ffn1_w = pr.next(HH*2*HH);lp[li].ffn1_b = pr.next(2*HH);
        lp[li].ffn2_w = pr.next(2*HH*HH);lp[li].ffn2_b = pr.next(HH);
        lp[li].ln2_g  = pr.next(HH);     lp[li].ln2_b  = pr.next(HH);
    }

    // ---------------- workspace layout ----------------
    float* ws = (float*)d_ws;
    const size_t NM = (size_t)MTOT * HH;   // 589824
    float* X  = ws;
    float* Q  = ws + 1 * NM;
    float* Km = ws + 2 * NM;
    float* V  = ws + 3 * NM;
    float* SK = ws + 4 * NM;
    float* H1 = ws + 5 * NM;
    float* H2 = ws + 6 * NM;
    _Float16* f16b = (_Float16*)(ws + 7 * NM);
    _Float16* X16  = f16b;                 // NM
    _Float16* SK16 = f16b + 1 * NM;        // NM
    _Float16* H116 = f16b + 2 * NM;        // NM
    _Float16* F16  = f16b + 3 * NM;        // 2*NM (ffn1 output, f16 only)
    _Float16* WPK  = f16b + 5 * NM;        // 3 * 147456 packed weights
    const size_t WL = 147456;              // per-layer packed weight elems

    const dim3 gBlk(256);
    const dim3 gGemm128(MTOT / 16, 1);     // P = 128
    const dim3 gGemm256(MTOT / 16, 2);     // P = 256

    // ---------------- pack weights ----------------
    _Float16* wp[3][7];
    for (int li = 0; li < 3; ++li) {
        _Float16* base = WPK + (size_t)li * WL;
        wp[li][0] = base;                      // q   16384
        wp[li][1] = base + 16384;              // k
        wp[li][2] = base + 2 * 16384;          // v
        wp[li][3] = base + 3 * 16384;          // skip
        wp[li][4] = base + 4 * 16384;          // O
        wp[li][5] = base + 5 * 16384;          // ffn1 32768
        wp[li][6] = base + 5 * 16384 + 32768;  // ffn2 32768
        pack_w_kernel<<<16384 / 256, gBlk, 0, stream>>>(lp[li].q_w,    wp[li][0], 128, 128);
        pack_w_kernel<<<16384 / 256, gBlk, 0, stream>>>(lp[li].k_w,    wp[li][1], 128, 128);
        pack_w_kernel<<<16384 / 256, gBlk, 0, stream>>>(lp[li].v_w,    wp[li][2], 128, 128);
        pack_w_kernel<<<16384 / 256, gBlk, 0, stream>>>(lp[li].skip_w, wp[li][3], 128, 128);
        pack_w_kernel<<<16384 / 256, gBlk, 0, stream>>>(lp[li].O_w,    wp[li][4], 128, 128);
        pack_w_kernel<<<32768 / 256, gBlk, 0, stream>>>(lp[li].ffn1_w, wp[li][5], 128, 256);
        pack_w_kernel<<<32768 / 256, gBlk, 0, stream>>>(lp[li].ffn2_w, wp[li][6], 256, 128);
    }

    // ---------------- forward ----------------
    embed_ln_kernel<<<MTOT, 128, 0, stream>>>(
        states, actions, rtg, tsteps,
        e_root_w, e_root_b, e_nr_w, e_nr_b, e_ret_w, e_ret_b, e_time,
        ln_g, ln_b, X, X16);

    for (int li = 0; li < 3; ++li) {
        const LayerP& P = lp[li];
        gemm_wmma<0, 128, 128><<<gGemm128, gBlk, 0, stream>>>(X16, wp[li][0], P.q_b,    nullptr, Q,  nullptr);
        gemm_wmma<0, 128, 128><<<gGemm128, gBlk, 0, stream>>>(X16, wp[li][1], P.k_b,    nullptr, Km, nullptr);
        gemm_wmma<0, 128, 128><<<gGemm128, gBlk, 0, stream>>>(X16, wp[li][2], P.v_b,    nullptr, V,  nullptr);
        gemm_wmma<0, 128, 128><<<gGemm128, gBlk, 0, stream>>>(X16, wp[li][3], P.skip_b, nullptr, SK, nullptr);

        attn_kernel<<<MTOT, 128, 0, stream>>>(Q, Km, V, SK, SK16);    // SK16 = agg + skip

        gemm_wmma<2, 128, 128><<<gGemm128, gBlk, 0, stream>>>(SK16, wp[li][4], P.O_b, X, H1, nullptr); // H1 = x + attn@O + b
        ln_kernel<<<MTOT, 128, 0, stream>>>(H1, P.ln1_g, P.ln1_b, H1, H116);

        gemm_wmma<1, 128, 256><<<gGemm256, gBlk, 0, stream>>>(H116, wp[li][5], P.ffn1_b, nullptr, nullptr, F16); // relu -> f16
        gemm_wmma<2, 256, 128><<<gGemm128, gBlk, 0, stream>>>(F16,  wp[li][6], P.ffn2_b, H1, H2, nullptr);       // h + ffn2
        ln_kernel<<<MTOT, 128, 0, stream>>>(H2, P.ln2_g, P.ln2_b, X, X16);
    }

    predict_kernel<<<B_ * L_ * 8, 128, 0, stream>>>(X, pred_w, pred_b, (float*)d_out);
}
